// VectorQuantizer_6760278524346
// MI455X (gfx1250) — matrix operand order repaired and checked by reading, not compiled, not run on hardware
//
#include <hip/hip_runtime.h>
#include <hip/hip_bf16.h>
#include <stdint.h>

typedef __attribute__((ext_vector_type(16))) __bf16 v16bf;
typedef __attribute__((ext_vector_type(8)))  float  v8f;

#define NUM_EMB        8192
#define EMB_DIM        64
#define N_ROWS         32768   // 32 * 32 * 32
#define KB             128     // codes staged into LDS per pipeline stage
#define NSTAGES        (NUM_EMB / KB)
#define ROWS_PER_BLOCK 128
#define THREADS        256

// ---------------------------------------------------------------------------
// Kernel 1: split codebook into bf16 hi/lo, stored in WMMA-B operand order,
// and compute per-code squared norms.
// B (32x16 bf16) k order used here (contiguous, same as A): lanes 0-15 hold K=0..15 for column
// n=lane (2 K-values per VGPR dword), lanes 16-31 hold K=16..31.
// ---------------------------------------------------------------------------
__global__ void vq_prep_kernel(const float* __restrict__ emb,
                               unsigned short* __restrict__ bhi,
                               unsigned short* __restrict__ blo,
                               float* __restrict__ cnorm) {
  const int k = blockIdx.x;    // code
  const int c = threadIdx.x;   // dim
  float e = emb[k * EMB_DIM + c];

  unsigned int u = __float_as_uint(e);
  unsigned short hi = (unsigned short)(u >> 16);          // truncate to bf16
  float hif = __uint_as_float(u & 0xFFFF0000u);
  float lo  = e - hif;                                    // exact residual
  unsigned short lob = (unsigned short)(__float_as_uint(lo) >> 16);

  const int t  = k >> 4;       // 16-code tile
  const int n  = k & 15;       // column within tile
  const int q  = c >> 5;       // 32-dim K chunk
  const int kk = c & 31;
  const int lane = n + ((kk & 16) ? 16 : 0);
  const int vg   = (kk & 15) >> 1;
  const int half = kk & 1;
  // each (tile,chunk) block = 32 lanes * 32 bytes = 1024 B = 512 ushorts
  const size_t idx = (size_t)(t * 2 + q) * 512 + (size_t)lane * 16 + vg * 2 + half;
  bhi[idx] = hi;
  blo[idx] = lob;

  __shared__ float s[EMB_DIM];
  s[c] = e * e;
  __syncthreads();
  if (c == 0) {
    float acc = 0.f;
#pragma unroll
    for (int i = 0; i < EMB_DIM; ++i) acc += s[i];
    cnorm[k] = acc;
  }
}

// ---------------------------------------------------------------------------
// Kernel 2: fused distance matmul (split-bf16 WMMA) + running argmin + gather,
// with double-buffered async global->LDS staging of the codebook tiles.
// ---------------------------------------------------------------------------
__global__ __launch_bounds__(THREADS) void vq_main_kernel(
    const float* __restrict__ hidden,
    const float* __restrict__ emb,
    const unsigned short* __restrict__ bhi,
    const unsigned short* __restrict__ blo,
    const float* __restrict__ cnorm,
    float* __restrict__ zq_out,
    int* __restrict__ idx_out) {

  __shared__ v16bf s_bhi[2][(KB / 16) * 2 * 32];  // 2 x 16 KB
  __shared__ v16bf s_blo[2][(KB / 16) * 2 * 32];  // 2 x 16 KB
  __shared__ float s_cn[KB];
  __shared__ int   s_idx[ROWS_PER_BLOCK];

  const int tid  = threadIdx.x;
  const int wave = tid >> 5;
  const int lane = tid & 31;
  const int m    = lane & 15;   // row-in-16 / code-in-16
  const int grp  = lane >> 4;

  const int rowsBase = blockIdx.x * ROWS_PER_BLOCK + wave * 16;
  const int nrow = rowsBase + m;
  const int bb   = nrow >> 10;         // image index (1024 pixels per image)
  const int pix  = nrow & 1023;
  const float* xp = hidden + (size_t)bb * 65536 + pix;  // stride over C = 1024

  // LDS byte offsets of the staging buffers (generic LDS addr bits [31:0]).
  const unsigned ldsH0 = (unsigned)(size_t)&s_bhi[0][0];
  const unsigned ldsH1 = (unsigned)(size_t)&s_bhi[1][0];
  const unsigned ldsL0 = (unsigned)(size_t)&s_blo[0][0];
  const unsigned ldsL1 = (unsigned)(size_t)&s_blo[1][0];

  // ---- Build A operands (16 rows x 64 dims, bf16 hi/lo split), per the
  //      same contiguous k order as B (k = 16*grp + s). ----
  unsigned short ah[2][16], al[2][16];
#pragma unroll
  for (int q = 0; q < 2; ++q) {
#pragma unroll
    for (int i = 0; i < 8; ++i) {
#pragma unroll
      for (int h = 0; h < 2; ++h) {
        int kk = 2 * i + h + (grp ? 16 : 0);
        int c  = q * 32 + kk;
        float xv = xp[(size_t)c * 1024];
        unsigned int u = __float_as_uint(xv);
        ah[q][i * 2 + h] = (unsigned short)(u >> 16);
        float hif = __uint_as_float(u & 0xFFFF0000u);
        float lo  = xv - hif;
        al[q][i * 2 + h] = (unsigned short)(__float_as_uint(lo) >> 16);
      }
    }
  }
  v16bf Ahi0, Ahi1, Alo0, Alo1;
  __builtin_memcpy(&Ahi0, ah[0], 32);
  __builtin_memcpy(&Ahi1, ah[1], 32);
  __builtin_memcpy(&Alo0, al[0], 32);
  __builtin_memcpy(&Alo1, al[1], 32);

  float minv[8];
  int   mini[8];
#pragma unroll
  for (int r = 0; r < 8; ++r) { minv[r] = 3.4e38f; mini[r] = 0; }

  // Issue the async copy of one 128-code stage into buffer `buf`.
  // Per thread: 4 x B128 (hi) + 4 x B128 (lo)  => ASYNCcnt += 8 per wave.
  auto issue_stage = [&](int kb, int buf) {
    const char* srcH = (const char*)bhi + (size_t)kb * 128;  // bytes
    const char* srcL = (const char*)blo + (size_t)kb * 128;
    unsigned dH = (buf ? ldsH1 : ldsH0);
    unsigned dL = (buf ? ldsL1 : ldsL0);
#pragma unroll
    for (int j = 0; j < (KB * 128 / 16) / THREADS; ++j) {    // 4 iters
      unsigned voff = (unsigned)((tid + j * THREADS) * 16);
      asm volatile("global_load_async_to_lds_b128 %0, %1, %2"
                   :: "v"(dH + voff), "v"(voff), "s"(srcH) : "memory");
      asm volatile("global_load_async_to_lds_b128 %0, %1, %2"
                   :: "v"(dL + voff), "v"(voff), "s"(srcL) : "memory");
    }
  };

  issue_stage(0, 0);   // prologue: stage 0 in flight

  for (int s = 0; s < NSTAGES; ++s) {
    const int kb  = s * KB;
    const int buf = s & 1;

    if (s + 1 < NSTAGES) {
      issue_stage(kb + KB, buf ^ 1);                       // prefetch next
      asm volatile("s_wait_asynccnt 0x8" ::: "memory");    // stage s done
    } else {
      asm volatile("s_wait_asynccnt 0x0" ::: "memory");
    }

    // Small cnorm stage (regular path; barrier-protected on both sides).
    for (int i = tid; i < KB; i += THREADS) s_cn[i] = cnorm[kb + i];
    __syncthreads();   // all waves' async + cnorm data visible

#pragma unroll 2
    for (int t = 0; t < KB / 16; ++t) {
      v16bf Bh0 = s_bhi[buf][(t * 2 + 0) * 32 + lane];
      v16bf Bh1 = s_bhi[buf][(t * 2 + 1) * 32 + lane];
      v16bf Bl0 = s_blo[buf][(t * 2 + 0) * 32 + lane];
      v16bf Bl1 = s_blo[buf][(t * 2 + 1) * 32 + lane];
      v8f acc = {};
      // dot = xhi.ehi + xlo.ehi + xhi.elo  (split-bf16, fp32 accumulate)
      acc = __builtin_amdgcn_wmma_f32_16x16x32_bf16(false, Ahi0, false, Bh0, (short)0, acc, false, false);
      acc = __builtin_amdgcn_wmma_f32_16x16x32_bf16(false, Ahi1, false, Bh1, (short)0, acc, false, false);
      acc = __builtin_amdgcn_wmma_f32_16x16x32_bf16(false, Alo0, false, Bh0, (short)0, acc, false, false);
      acc = __builtin_amdgcn_wmma_f32_16x16x32_bf16(false, Alo1, false, Bh1, (short)0, acc, false, false);
      acc = __builtin_amdgcn_wmma_f32_16x16x32_bf16(false, Ahi0, false, Bl0, (short)0, acc, false, false);
      acc = __builtin_amdgcn_wmma_f32_16x16x32_bf16(false, Ahi1, false, Bl1, (short)0, acc, false, false);

      float cn  = s_cn[t * 16 + m];
      int  code = kb + t * 16 + m;
#pragma unroll
      for (int r = 0; r < 8; ++r) {
        float d = __builtin_fmaf(-2.0f, acc[r], cn);  // ||e||^2 - 2 x.e
        if (d < minv[r]) { minv[r] = d; mini[r] = code; }
      }
    }
    __syncthreads();   // done reading buf before it is overwritten (s+2)
  }

  // Butterfly min-reduce across the 16 lanes holding each row's codes.
#pragma unroll
  for (int off = 1; off < 16; off <<= 1) {
#pragma unroll
    for (int r = 0; r < 8; ++r) {
      float ov = __shfl_xor(minv[r], off, 32);
      int   oi = __shfl_xor(mini[r], off, 32);
      if (ov < minv[r] || (ov == minv[r] && oi < mini[r])) {
        minv[r] = ov;
        mini[r] = oi;
      }
    }
  }

  if (m == 0) {
#pragma unroll
    for (int r = 0; r < 8; ++r) {
      int row = r + (grp ? 8 : 0);       // C/D layout: lanes 16-31 hold M=8..15
      s_idx[wave * 16 + row] = mini[r];
      idx_out[rowsBase + row] = mini[r];
    }
  }
  __syncthreads();

  // Gather z_q = emb[idx] back into NCHW (fp32 exact), coalesced over pixels.
  const int pixBase = (blockIdx.x & 7) * ROWS_PER_BLOCK;  // 8 blocks per image
  const int bimg    = blockIdx.x >> 3;
  for (int v = tid; v < ROWS_PER_BLOCK * EMB_DIM; v += THREADS) {
    int mrow = v & (ROWS_PER_BLOCK - 1);
    int c    = v >> 7;
    int k    = s_idx[mrow];
    zq_out[(size_t)bimg * 65536 + (size_t)c * 1024 + pixBase + mrow] =
        emb[k * EMB_DIM + c];
  }
}

// ---------------------------------------------------------------------------
extern "C" void kernel_launch(void* const* d_in, const int* in_sizes, int n_in,
                              void* d_out, int out_size, void* d_ws, size_t ws_size,
                              hipStream_t stream) {
  (void)in_sizes; (void)n_in; (void)out_size; (void)ws_size;
  const float* hidden = (const float*)d_in[0];
  const float* emb    = (const float*)d_in[1];

  float* zq      = (float*)d_out;
  int*   idx_out = (int*)d_out + (size_t)N_ROWS * EMB_DIM;  // after z_q

  // d_ws layout: bhi (1 MB) | blo (1 MB) | cnorm (32 KB)
  unsigned short* bhi = (unsigned short*)d_ws;
  unsigned short* blo = bhi + (size_t)NUM_EMB * EMB_DIM;
  float* cnorm        = (float*)(blo + (size_t)NUM_EMB * EMB_DIM);

  vq_prep_kernel<<<NUM_EMB, EMB_DIM, 0, stream>>>(emb, bhi, blo, cnorm);
  vq_main_kernel<<<N_ROWS / ROWS_PER_BLOCK, THREADS, 0, stream>>>(
      hidden, emb, bhi, blo, cnorm, zq, idx_out);
}
